// SparseMHADecoder_59974923321649
// MI455X (gfx1250) — matrix operand order repaired and checked by reading, not compiled, not run on hardware
//
#include <hip/hip_runtime.h>
#include <hip/hip_bf16.h>
#include <stdint.h>

typedef __attribute__((ext_vector_type(16))) __bf16 v16bf;
typedef __attribute__((ext_vector_type(8)))  float  v8f;

#define B_   2
#define LQ_  2048
#define LKV_ 512
#define H_   16
#define DHID 1024

#if defined(__HIP_DEVICE_COMPILE__) && __has_builtin(__builtin_amdgcn_tensor_load_to_lds)
#define USE_TDM 1
#else
#define USE_TDM 0
#endif

// ---------------------------------------------------------------------------
// Fragment loaders for 16-bit WMMA operands, contiguous k order k = hi*16 + s in A and B.
// A (16x32, row-major, lane m = lane&15, hi = lane>>4):
//   K elements {hi*16 .. hi*16+15}                           (one 32B span)
// B (32x16) loaded from an n-major ("transposed") buffer Bt[n][k]:
//   lane n = lane&15, K elements {hi*16 .. hi*16+15}          (one 32B span)
// ---------------------------------------------------------------------------
__device__ inline v16bf frag_a_ld(const __bf16* p, int ld) {
  const int lane = threadIdx.x & 31;
  const int m = lane & 15, hi = lane >> 4;
  const __bf16* r0 = p + (size_t)m * ld + hi * 16;
  v16bf f;
#pragma unroll
  for (int i = 0; i < 8; ++i) f[i] = r0[i];
#pragma unroll
  for (int i = 0; i < 8; ++i) f[i + 8] = r0[8 + i];
  return f;
}

__device__ inline v16bf frag_bt_ld(const __bf16* p, int ld) {
  const int lane = threadIdx.x & 31;
  const int n = lane & 15, hi = lane >> 4;
  const __bf16* r0 = p + (size_t)n * ld + hi * 16;
  v16bf f;
#pragma unroll
  for (int i = 0; i < 16; ++i) f[i] = r0[i];
  return f;
}

#if USE_TDM
typedef unsigned int u32x4 __attribute__((ext_vector_type(4)));
typedef int          i32x8 __attribute__((ext_vector_type(8)));
typedef int          i32x4 __attribute__((ext_vector_type(4)));

__device__ inline uint32_t lds_off(void* p) {
  return (uint32_t)(uintptr_t)(__attribute__((address_space(3))) void*)p;
}

// TDM 2D tile load: tile_w x tile_h bf16 elements from a row-major tensor with
// row stride `stride_e` elements into LDS at byte address `lds_b`, padding the
// LDS destination with 4 DWORDs (16B) every 16 DWORDs (64B) -> 40-halfword rows.
// D# layout per CDNA5 ISA ch. 8 (groups 0/1; groups 2/3 zero for 2D tiles).
__device__ inline void tdm_load_tile_2d(const __bf16* gptr, uint32_t lds_b,
                                        int tensor_w, int tile_w, int tile_h,
                                        int stride_e) {
  const uint64_t ga = (uint64_t)(uintptr_t)gptr;
  u32x4 g0;
  g0[0] = 1u;                                    // count=1 (valid), user mode
  g0[1] = lds_b;                                 // lds_addr [63:32]
  g0[2] = (uint32_t)ga;                          // global_addr lo
  g0[3] = (uint32_t)((ga >> 32) & 0x1FFFFFFu)    // global_addr [56:32]
        | (2u << 30);                            // type=2 ("image") [127:126]
  i32x8 g1;
  g1[0] = (int)((1u << 16)                       // data_size = 1 -> 2 bytes
              | (1u << 20)                       // pad_enable
              | (3u << 22)                       // pad_interval: 16 DWORDs
              | (3u << 25));                     // pad_amount:   4 DWORDs
  g1[1] = (int)(((uint32_t)tensor_w & 0xFFFFu) << 16);            // dim0[15:0]
  g1[2] = (int)((((uint32_t)tensor_w >> 16) & 0xFFFFu)            // dim0[31:16]
              | (((uint32_t)tile_h & 0xFFFFu) << 16));            // dim1[15:0]
  g1[3] = (int)(((uint32_t)tile_w & 0xFFFFu) << 16);              // tile_dim0
  g1[4] = (int)((uint32_t)tile_h & 0xFFFFu);                      // tile_dim1
  g1[5] = (int)(uint32_t)stride_e;                                // dim0_stride lo32
  g1[6] = 0;
  g1[7] = 0;
  i32x4 z4; z4[0] = 0; z4[1] = 0; z4[2] = 0; z4[3] = 0;
  i32x8 z8;
#pragma unroll
  for (int i = 0; i < 8; ++i) z8[i] = 0;
  // 6-arg form (clang-23 / therock-10.0 arity).
  __builtin_amdgcn_tensor_load_to_lds(g0, g1, z4, z4, z8, 0);
}
#endif

// ---------------------------------------------------------------------------
// f32 -> bf16 elementwise convert
// ---------------------------------------------------------------------------
__global__ __launch_bounds__(256) void cvt_f32_bf16(const float* __restrict__ s,
                                                    __bf16* __restrict__ d, int n) {
  int i = blockIdx.x * 256 + threadIdx.x;
  if (i < n) d[i] = (__bf16)s[i];
}

// f32 [K][N] -> bf16 transposed [N][K] (32x32 LDS tiles)
__global__ __launch_bounds__(256) void cvt_transpose(const float* __restrict__ src,
                                                     __bf16* __restrict__ dst,
                                                     int K, int N) {
  __shared__ float tile[32][33];
  const int k0 = blockIdx.y * 32, n0 = blockIdx.x * 32;
#pragma unroll
  for (int i = threadIdx.y; i < 32; i += 8)
    tile[i][threadIdx.x] = src[(size_t)(k0 + i) * N + n0 + threadIdx.x];
  __syncthreads();
#pragma unroll
  for (int i = threadIdx.y; i < 32; i += 8)
    dst[(size_t)(n0 + i) * K + k0 + threadIdx.x] = (__bf16)tile[threadIdx.x][i];
}

// ---------------------------------------------------------------------------
// bf16 GEMM: C[M][N] = A[M][K] * B[K][N], with B supplied transposed BT[N][K].
// 256 threads = 8 waves, block tile 128x128, K-step 32, each wave -> 32x64.
// Tiles staged into double-buffered LDS by the Tensor Data Mover (wave 0
// issues descriptors, s_wait_tensorcnt pipelines two tile-pairs in flight).
// ---------------------------------------------------------------------------
__global__ __launch_bounds__(256) void gemm_bf16_nt(const __bf16* __restrict__ A,
                                                    const __bf16* __restrict__ BT,
                                                    void* __restrict__ C,
                                                    int M, int N, int K, int out_f32) {
  __shared__ __bf16 As[2][128 * 40];
  __shared__ __bf16 Bs[2][128 * 40];

  const int t = threadIdx.x;
  const int m0 = blockIdx.y * 128, n0 = blockIdx.x * 128;
  const int wave = t >> 5, lane = t & 31;
  const int wm = wave >> 1, wn = wave & 1;

  v8f acc[2][4];
#pragma unroll
  for (int i = 0; i < 2; ++i)
#pragma unroll
    for (int j = 0; j < 4; ++j)
#pragma unroll
      for (int r = 0; r < 8; ++r) acc[i][j][r] = 0.f;

#if USE_TDM
  if (wave == 0) {  // prologue: stage k-tile 0 into buffer 0
    tdm_load_tile_2d(A  + (size_t)m0 * K, lds_off(As[0]), K, 32, 128, K);
    tdm_load_tile_2d(BT + (size_t)n0 * K, lds_off(Bs[0]), K, 32, 128, K);
  }
#else
  const int lr = t >> 1;        // 0..127: tile row this thread fills
  const int lc = (t & 1) * 16;  // 0 or 16: K half
  {
    const uint4* sa = (const uint4*)(A + (size_t)(m0 + lr) * K + lc);
    uint4* da = (uint4*)(As[0] + lr * 40 + lc);
    da[0] = sa[0]; da[1] = sa[1];
    const uint4* sb = (const uint4*)(BT + (size_t)(n0 + lr) * K + lc);
    uint4* db = (uint4*)(Bs[0] + lr * 40 + lc);
    db[0] = sb[0]; db[1] = sb[1];
  }
#endif

  int cur = 0;
  for (int kt = 0; kt < K; kt += 32) {
#if USE_TDM
    if (wave == 0) {
      if (kt + 32 < K) {  // issue next tile-pair into the other buffer
        tdm_load_tile_2d(A  + (size_t)m0 * K + kt + 32, lds_off(As[cur ^ 1]),
                         K, 32, 128, K);
        tdm_load_tile_2d(BT + (size_t)n0 * K + kt + 32, lds_off(Bs[cur ^ 1]),
                         K, 32, 128, K);
        __builtin_amdgcn_s_wait_tensorcnt(2);  // oldest pair (cur) complete
      } else {
        __builtin_amdgcn_s_wait_tensorcnt(0);
      }
    }
#else
    if (kt + 32 < K) {
      const uint4* sa = (const uint4*)(A + (size_t)(m0 + lr) * K + kt + 32 + lc);
      uint4* da = (uint4*)(As[cur ^ 1] + lr * 40 + lc);
      da[0] = sa[0]; da[1] = sa[1];
      const uint4* sb = (const uint4*)(BT + (size_t)(n0 + lr) * K + kt + 32 + lc);
      uint4* db = (uint4*)(Bs[cur ^ 1] + lr * 40 + lc);
      db[0] = sb[0]; db[1] = sb[1];
      __builtin_prefetch(A + (size_t)(m0 + lr) * K + kt + 64 + lc, 0, 0);
      __builtin_prefetch(BT + (size_t)(n0 + lr) * K + kt + 64 + lc, 0, 0);
    }
#endif
    __syncthreads();  // current buffer ready for all waves

    v16bf a0 = frag_a_ld(As[cur] + (wm * 32) * 40, 40);
    v16bf a1 = frag_a_ld(As[cur] + (wm * 32 + 16) * 40, 40);
#pragma unroll
    for (int j = 0; j < 4; ++j) {
      v16bf bj = frag_bt_ld(Bs[cur] + (wn * 64 + j * 16) * 40, 40);
      acc[0][j] = __builtin_amdgcn_wmma_f32_16x16x32_bf16(
          false, a0, false, bj, (short)0, acc[0][j], false, false);
      acc[1][j] = __builtin_amdgcn_wmma_f32_16x16x32_bf16(
          false, a1, false, bj, (short)0, acc[1][j], false, false);
    }
    __syncthreads();  // all reads of `cur` done before it is refilled
    cur ^= 1;
  }

  const int n = lane & 15, hi = lane >> 4;
#pragma unroll
  for (int i = 0; i < 2; ++i)
#pragma unroll
    for (int j = 0; j < 4; ++j)
#pragma unroll
      for (int r = 0; r < 8; ++r) {
        const int row = m0 + wm * 32 + i * 16 + hi * 8 + r;
        const int col = n0 + wn * 64 + j * 16 + n;
        if (out_f32) ((float*)C)[(size_t)row * N + col] = acc[i][j][r];
        else         ((__bf16*)C)[(size_t)row * N + col] = (__bf16)acc[i][j][r];
      }
}

// ---------------------------------------------------------------------------
// Banded sparse attention. One wave per (b, h, 16-query tile).
// Valid keys for query j:  kv in [max(0,ceil((j-127)/4)), j>>2]  (<= 32 keys).
// 48-key window starting at kw0 = j0/4 - 31 covers the whole 16-query tile.
//   S^T = K_win(48x64) x Q^T(64x16)  -> WMMA, masked softmax in LDS,
//   O   = P(16x64,pad) x V_win       -> WMMA (V staged transposed in LDS).
// ---------------------------------------------------------------------------
__global__ __launch_bounds__(128) void attn_kernel(const __bf16* __restrict__ QP,
                                                   const __bf16* __restrict__ KP,
                                                   const __bf16* __restrict__ VP,
                                                   __bf16* __restrict__ OB) {
  __shared__ __bf16 Vt[4][64 * 64];   // [d][kv], kv padded/zeroed to 64
  __shared__ float  Sl[4][16 * 48];   // [q][kv_local]
  __shared__ __bf16 Pl[4][16 * 64];   // [q][kv_local], zero-padded

  const int wave = threadIdx.x >> 5;
  const int lane = threadIdx.x & 31;
  const int u   = blockIdx.x * 4 + wave;   // 0 .. B*H*(LQ/16)-1 = 4095
  const int b   = u >> 11;
  const int rem = u & 2047;
  const int h   = rem >> 7;
  const int j0  = (rem & 127) << 4;
  const int kw0 = (j0 >> 2) - 31;

  __bf16* vt = Vt[wave];
  float*  sl = Sl[wave];
  __bf16* pl = Pl[wave];

  // ---- stage V window transposed into LDS: vt[d*64 + kl] ----
  for (int kl = lane; kl < 48; kl += 32) {
    int kv = kw0 + kl;
    int c  = kv < 0 ? 0 : (kv > 511 ? 511 : kv);
    const __bf16* src = VP + (size_t)(b * LKV_ + c) * DHID + h * 64;
    __builtin_prefetch(KP + (size_t)(b * LKV_ + c) * DHID + h * 64, 0, 0);
#pragma unroll
    for (int d = 0; d < 64; ++d) vt[d * 64 + kl] = src[d];
  }
  for (int idx = lane; idx < 64 * 16; idx += 32) {   // zero pad kv 48..63
    int d = idx >> 4, kz = idx & 15;
    vt[d * 64 + 48 + kz] = (__bf16)0.f;
  }
  __syncthreads();

  // ---- scores S^T via WMMA: 3 key-chunks x (K=64 in 2 steps) ----
  {
    const int m = lane & 15, hi = lane >> 4;
#pragma unroll
    for (int mt = 0; mt < 3; ++mt) {
      int kv = kw0 + mt * 16 + m;
      int c  = kv < 0 ? 0 : (kv > 511 ? 511 : kv);
      const __bf16* krow = KP + (size_t)(b * LKV_ + c) * DHID + h * 64;       // A: key rows
      const __bf16* qcol = QP + (size_t)(b * LQ_ + j0 + m) * DHID + h * 64;   // B: query cols
      v8f acc;
#pragma unroll
      for (int r = 0; r < 8; ++r) acc[r] = 0.f;
#pragma unroll
      for (int ks = 0; ks < 2; ++ks) {
        v16bf af, bf_;
        const __bf16* ar = krow + ks * 32 + hi * 16;
#pragma unroll
        for (int i = 0; i < 8; ++i) { af[i] = ar[i]; af[i + 8] = ar[8 + i]; }
        const __bf16* br = qcol + ks * 32 + hi * 16;
#pragma unroll
        for (int i = 0; i < 16; ++i) bf_[i] = br[i];
        acc = __builtin_amdgcn_wmma_f32_16x16x32_bf16(
            false, af, false, bf_, (short)0, acc, false, false);
      }
#pragma unroll
      for (int r = 0; r < 8; ++r)                       // transpose into S[q][kv]
        sl[m * 48 + mt * 16 + hi * 8 + r] = acc[r] * 0.125f;   // 1/sqrt(64)
    }
  }
  __syncthreads();

  // ---- masked softmax, one query per lane (lanes 0..15) ----
  if (lane < 16) {
    const int j  = j0 + lane;
    int lo = (j - 124) >> 2;  if (lo < 0) lo = 0;   // ceil((j-127)/4)
    const int hiK = j >> 2;                         // <= 511 since j <= 2047
    float mx = -3.0e38f;
    for (int kl = 0; kl < 48; ++kl) {
      int kv = kw0 + kl;
      float s = (kv >= lo && kv <= hiK) ? sl[lane * 48 + kl] : -3.0e38f;
      mx = s > mx ? s : mx;
    }
    float sum = 0.f;
    for (int kl = 0; kl < 48; ++kl) {
      int kv = kw0 + kl;
      float p = (kv >= lo && kv <= hiK) ? __expf(sl[lane * 48 + kl] - mx) : 0.f;
      sum += p;
      sl[lane * 48 + kl] = p;
    }
    const float inv = 1.f / sum;
    for (int kl = 0; kl < 48; ++kl) pl[lane * 64 + kl] = (__bf16)(sl[lane * 48 + kl] * inv);
    for (int kl = 48; kl < 64; ++kl) pl[lane * 64 + kl] = (__bf16)0.f;
  }
  __syncthreads();

  // ---- O = P x V via WMMA: 4 d-chunks x (K=64 in 2 steps) ----
  {
    const int n = lane & 15, hi = lane >> 4;
#pragma unroll
    for (int nt = 0; nt < 4; ++nt) {
      v8f acc;
#pragma unroll
      for (int r = 0; r < 8; ++r) acc[r] = 0.f;
#pragma unroll
      for (int kc = 0; kc < 2; ++kc) {
        v16bf af  = frag_a_ld(pl + kc * 32, 64);
        v16bf bf_ = frag_bt_ld(vt + (nt * 16) * 64 + kc * 32, 64);
        acc = __builtin_amdgcn_wmma_f32_16x16x32_bf16(
            false, af, false, bf_, (short)0, acc, false, false);
      }
#pragma unroll
      for (int r = 0; r < 8; ++r) {
        const int j = j0 + hi * 8 + r;
        OB[(size_t)(b * LQ_ + j) * DHID + h * 64 + nt * 16 + n] = (__bf16)acc[r];
      }
    }
  }
}

// ---------------------------------------------------------------------------
extern "C" void kernel_launch(void* const* d_in, const int* in_sizes, int n_in,
                              void* d_out, int out_size, void* d_ws, size_t ws_size,
                              hipStream_t stream) {
  (void)in_sizes; (void)n_in; (void)out_size; (void)ws_size;
  const float* q  = (const float*)d_in[0];
  const float* k  = (const float*)d_in[1];
  const float* v  = (const float*)d_in[2];
  const float* Wq = (const float*)d_in[3];
  const float* Wk = (const float*)d_in[4];
  const float* Wv = (const float*)d_in[5];
  const float* Wo = (const float*)d_in[6];
  float* out = (float*)d_out;

  __bf16* ws = (__bf16*)d_ws;
  const size_t M1 = (size_t)1 << 20;
  __bf16* qb  = ws + 0  * M1;  // 4M : q bf16       [4096][1024]
  __bf16* kb  = ws + 4  * M1;  // 1M : k bf16       [1024][1024]
  __bf16* vb  = ws + 5  * M1;  // 1M : v bf16
  __bf16* WqT = ws + 6  * M1;  // 1M : Wq^T bf16    [N][K]
  __bf16* WkT = ws + 7  * M1;
  __bf16* WvT = ws + 8  * M1;
  __bf16* WoT = ws + 9  * M1;
  __bf16* QPp = ws + 10 * M1;  // 4M : Q proj bf16  [4096][1024]
  __bf16* KPp = ws + 14 * M1;  // 1M
  __bf16* VPp = ws + 15 * M1;  // 1M
  __bf16* OBp = ws + 16 * M1;  // 4M : attn out bf16 -> 20M total

  const int nq = B_ * LQ_ * DHID;   // 4Mi
  const int nk = B_ * LKV_ * DHID;  // 1Mi
  cvt_f32_bf16<<<(nq + 255) / 256, 256, 0, stream>>>(q, qb, nq);
  cvt_f32_bf16<<<(nk + 255) / 256, 256, 0, stream>>>(k, kb, nk);
  cvt_f32_bf16<<<(nk + 255) / 256, 256, 0, stream>>>(v, vb, nk);

  dim3 tb(32, 8), tg(32, 32);
  cvt_transpose<<<tg, tb, 0, stream>>>(Wq, WqT, 1024, 1024);
  cvt_transpose<<<tg, tb, 0, stream>>>(Wk, WkT, 1024, 1024);
  cvt_transpose<<<tg, tb, 0, stream>>>(Wv, WvT, 1024, 1024);
  cvt_transpose<<<tg, tb, 0, stream>>>(Wo, WoT, 1024, 1024);

  gemm_bf16_nt<<<dim3(8, 32), 256, 0, stream>>>(qb, WqT, QPp, 4096, 1024, 1024, 0);
  gemm_bf16_nt<<<dim3(8, 8),  256, 0, stream>>>(kb, WkT, KPp, 1024, 1024, 1024, 0);
  gemm_bf16_nt<<<dim3(8, 8),  256, 0, stream>>>(vb, WvT, VPp, 1024, 1024, 1024, 0);

  attn_kernel<<<1024, 128, 0, stream>>>(QPp, KPp, VPp, OBp);

  gemm_bf16_nt<<<dim3(8, 32), 256, 0, stream>>>(OBp, WoT, out, 4096, 1024, 1024, 1);
}
